// Attn_35407710389025
// MI455X (gfx1250) — matrix operand order repaired and checked by reading, not compiled, not run on hardware
//
#include <hip/hip_runtime.h>

typedef __attribute__((ext_vector_type(16))) __bf16 v16bf;
typedef __attribute__((ext_vector_type(8)))  float  v8f;
typedef unsigned short u16;
typedef unsigned int   u32;

#define TT  2048
#define CC  768
#define NHH 12
#define HD  64

__device__ __forceinline__ u16 f2bf(float f) {
  u32 u = __float_as_uint(f);
  u32 r = u + 0x7fffu + ((u >> 16) & 1u);   // round-to-nearest-even
  return (u16)(r >> 16);
}

union BF16x16 { v16bf v; uint4 q[2]; };

// Build a v16bf A/B operand from two 16-byte (8 x bf16) aligned loads.
__device__ __forceinline__ v16bf load_tile16(const u16* p0, const u16* p1) {
  BF16x16 t;
  t.q[0] = *(const uint4*)p0;
  t.q[1] = *(const uint4*)p1;
  return t.v;
}

__device__ __forceinline__ v16bf ones16() {
  BF16x16 t;
  t.q[0] = make_uint4(0x3f803f80u, 0x3f803f80u, 0x3f803f80u, 0x3f803f80u);
  t.q[1] = t.q[0];
  return t.v;
}

// ---------------------------------------------------------------- convert
__global__ void k_convert(const float* __restrict__ in, u16* __restrict__ out, int n) {
  int i = blockIdx.x * blockDim.x + threadIdx.x;
  int stride = gridDim.x * blockDim.x;
  for (; i < n; i += stride) out[i] = f2bf(in[i]);
}

// ------------------------------------------------- QKV projection + RoPE
// qkv[b,t,d] = sum_c x[b,t,c] * w_qkv[d,c]; one wave = 16 rows x 128 cols.
// 128 | 768 so a col-tile never crosses the q/k/v boundary; it spans 2 heads.
__global__ __launch_bounds__(128) void k_qkv(const u16* __restrict__ xb,
                                             const u16* __restrict__ wb,
                                             u16* __restrict__ Qd,
                                             u16* __restrict__ Kd,
                                             u16* __restrict__ Vt) {
  const int lane = threadIdx.x & 31;
  const int half = lane >> 4;
  const int n    = lane & 15;
  const int gid  = blockIdx.x * 4 + (threadIdx.x >> 5);
  const int MT   = (4 * TT) / 16;           // 512 row tiles
  const int mtile = gid % MT;
  const int ntile = gid / MT;               // 0..17
  const int m0 = mtile * 16;
  const int b  = m0 / TT;
  const int t0 = m0 % TT;
  const int nb = ntile * 128;
  const int which = nb / CC;                // 0=q 1=k 2=v (constant per tile)
  const int h0 = ((ntile % 6) * 2);         // first head in tile

  v8f acc[8] = {};
  const u16* arow = xb + (size_t)(m0 + n) * CC;       // A row = lane&15
  for (int k0 = 0; k0 < CC; k0 += 32) {
    v16bf a = load_tile16(arow + k0 + half * 16, arow + k0 + half * 16 + 8);
#pragma unroll
    for (int s = 0; s < 8; ++s) {
      const u16* brow = wb + (size_t)(nb + s * 16 + n) * CC + k0 + half * 16;
      v16bf bt = load_tile16(brow, brow + 8);
      acc[s] = __builtin_amdgcn_wmma_f32_16x16x32_bf16(
          false, a, false, bt, (short)0, acc[s], false, false);
    }
  }

  // RoPE on d<16 of each head (subtiles 0 and 4) for q and k.
  if (which < 2) {
#pragma unroll
    for (int s = 0; s < 8; s += 4) {
      float rot[8];
#pragma unroll
      for (int r = 0; r < 8; ++r) {
        float other = __shfl_xor(acc[s][r], 8, 32);
        int   t  = t0 + r + 8 * half;
        int   fi = n & 7;
        float freq = __expf(-(float)fi * 1.1512925465f);  // 10000^(-fi/8)
        float ang  = (float)t * freq;
        float sn, cs;
        __sincosf(ang, &sn, &cs);
        float v = acc[s][r];
        rot[r] = (n < 8) ? (v * cs - other * sn) : (v * cs + other * sn);
      }
#pragma unroll
      for (int r = 0; r < 8; ++r) acc[s][r] = rot[r];
    }
  }

#pragma unroll
  for (int s = 0; s < 8; ++s) {
    const int h  = h0 + (s >> 2);
    const int bh = b * NHH + h;
    const int d  = (s & 3) * 16 + n;
#pragma unroll
    for (int r = 0; r < 8; ++r) {
      int t = t0 + r + 8 * half;
      u16 bv = f2bf(acc[s][r]);
      if (which == 0)      Qd[((size_t)bh * TT + t) * HD + d] = bv;
      else if (which == 1) Kd[((size_t)bh * TT + t) * HD + d] = bv;
      else                 Vt[((size_t)bh * HD + d) * TT + t] = bv;  // V transposed
    }
  }
}

// --------------------------------------------------- flash attention core
// One wave handles a 16-row Q tile; iterates key blocks of 64 with online
// softmax. Row sums computed via P x ones WMMA (no shuffle reduction).
__global__ __launch_bounds__(128) void k_attn(const u16* __restrict__ Q,
                                              const u16* __restrict__ K,
                                              const u16* __restrict__ V,
                                              u16* __restrict__ O) {
  __shared__ __align__(16) u16 smem[4][16 * 64];
  const int lane = threadIdx.x & 31;
  const int half = lane >> 4;
  const int n    = lane & 15;
  const int w    = threadIdx.x >> 5;
  const int gid  = blockIdx.x * 4 + w;
  const int qt = gid % (TT / 16);
  const int bh = gid / (TT / 16);           // 0..47
  const int b  = bh / NHH;
  const int h  = bh % NHH;
  const int qb = qt * 16;

  const u16* Qb = Q + (size_t)bh * TT * HD;
  const u16* Kb = K + (size_t)bh * TT * HD;
  const u16* Vb = V + (size_t)bh * HD * TT;

  v16bf aq[2];
#pragma unroll
  for (int c = 0; c < 2; ++c) {
    const u16* p = Qb + (size_t)(qb + n) * HD + c * 32 + half * 16;
    aq[c] = load_tile16(p, p + 8);
  }
  const v16bf bones = ones16();

  float m[8], l[8];
#pragma unroll
  for (int r = 0; r < 8; ++r) { m[r] = -3.0e38f; l[r] = 0.f; }
  v8f acc[4] = {};

  u16* sp = &smem[w][0];

  for (int kb = 0; kb <= qb + 15; kb += 64) {
    // S = Q.K^T for 4 key subtiles of 16 (keys kb .. kb+63)
    v8f s[4] = {};
#pragma unroll
    for (int c = 0; c < 2; ++c) {
#pragma unroll
      for (int j = 0; j < 4; ++j) {
        const u16* p = Kb + (size_t)(kb + j * 16 + n) * HD + c * 32 + half * 16;
        v16bf bk = load_tile16(p, p + 8);
        s[j] = __builtin_amdgcn_wmma_f32_16x16x32_bf16(
            false, aq[c], false, bk, (short)0, s[j], false, false);
      }
    }

    float alpha[8];
#pragma unroll
    for (int r = 0; r < 8; ++r) {
      int q = qb + r + 8 * half;
      float v0 = (kb      + n <= q) ? s[0][r] * 0.125f : -3.0e38f;
      float v1 = (kb + 16 + n <= q) ? s[1][r] * 0.125f : -3.0e38f;
      float v2 = (kb + 32 + n <= q) ? s[2][r] * 0.125f : -3.0e38f;
      float v3 = (kb + 48 + n <= q) ? s[3][r] * 0.125f : -3.0e38f;
      float mx = fmaxf(fmaxf(v0, v1), fmaxf(v2, v3));
      mx = fmaxf(mx, __shfl_xor(mx, 1, 32));
      mx = fmaxf(mx, __shfl_xor(mx, 2, 32));
      mx = fmaxf(mx, __shfl_xor(mx, 4, 32));
      mx = fmaxf(mx, __shfl_xor(mx, 8, 32));
      float mn = fmaxf(m[r], mx);
      alpha[r] = __expf(m[r] - mn);
      m[r] = mn;
      float p0 = __expf(v0 - mn);
      float p1 = __expf(v1 - mn);
      float p2 = __expf(v2 - mn);
      float p3 = __expf(v3 - mn);
#pragma unroll
      for (int ss = 0; ss < 4; ++ss) acc[ss][r] *= alpha[r];
      int row = (r + 8 * half) * 64;
      sp[row + n]      = f2bf(p0);
      sp[row + 16 + n] = f2bf(p1);
      sp[row + 32 + n] = f2bf(p2);
      sp[row + 48 + n] = f2bf(p3);
    }

    // Read P back in A layout: chunk 0 = keys 0..31, chunk 1 = keys 32..63
    const u16* pr = sp + n * 64 + half * 16;
    v16bf ap0 = load_tile16(pr,      pr + 8);
    v16bf ap1 = load_tile16(pr + 32, pr + 40);

    // Row sums via P x ones (lands in C/D layout, replicated over columns)
    v8f rs = {};
    rs = __builtin_amdgcn_wmma_f32_16x16x32_bf16(
        false, ap0, false, bones, (short)0, rs, false, false);
    rs = __builtin_amdgcn_wmma_f32_16x16x32_bf16(
        false, ap1, false, bones, (short)0, rs, false, false);
#pragma unroll
    for (int r = 0; r < 8; ++r) l[r] = l[r] * alpha[r] + rs[r];

    // O += P.V
#pragma unroll
    for (int ss = 0; ss < 4; ++ss) {
      const u16* vp0 = Vb + (size_t)(ss * 16 + n) * TT + kb + half * 16;
      v16bf bv0 = load_tile16(vp0, vp0 + 8);
      acc[ss] = __builtin_amdgcn_wmma_f32_16x16x32_bf16(
          false, ap0, false, bv0, (short)0, acc[ss], false, false);
      const u16* vp1 = vp0 + 32;
      v16bf bv1 = load_tile16(vp1, vp1 + 8);
      acc[ss] = __builtin_amdgcn_wmma_f32_16x16x32_bf16(
          false, ap1, false, bv1, (short)0, acc[ss], false, false);
    }
  }

#pragma unroll
  for (int ss = 0; ss < 4; ++ss) {
#pragma unroll
    for (int r = 0; r < 8; ++r) {
      int t   = qb + r + 8 * half;
      int col = h * HD + ss * 16 + n;
      O[((size_t)b * TT + t) * CC + col] = f2bf(acc[ss][r] / l[r]);
    }
  }
}

// ------------------------------------------------------ output projection
__global__ __launch_bounds__(128) void k_proj(const u16* __restrict__ Ob,
                                              const u16* __restrict__ wb,
                                              float* __restrict__ out) {
  const int lane = threadIdx.x & 31;
  const int half = lane >> 4;
  const int n    = lane & 15;
  const int gid  = blockIdx.x * 4 + (threadIdx.x >> 5);
  const int MT   = (4 * TT) / 16;
  const int mtile = gid % MT;
  const int ntile = gid / MT;               // 0..5
  const int m0 = mtile * 16;
  const int nb = ntile * 128;

  v8f acc[8] = {};
  const u16* arow = Ob + (size_t)(m0 + n) * CC;
  for (int k0 = 0; k0 < CC; k0 += 32) {
    v16bf a = load_tile16(arow + k0 + half * 16, arow + k0 + half * 16 + 8);
#pragma unroll
    for (int s = 0; s < 8; ++s) {
      const u16* brow = wb + (size_t)(nb + s * 16 + n) * CC + k0 + half * 16;
      v16bf bt = load_tile16(brow, brow + 8);
      acc[s] = __builtin_amdgcn_wmma_f32_16x16x32_bf16(
          false, a, false, bt, (short)0, acc[s], false, false);
    }
  }
#pragma unroll
  for (int s = 0; s < 8; ++s) {
#pragma unroll
    for (int r = 0; r < 8; ++r) {
      int row = m0 + r + 8 * half;
      int col = nb + s * 16 + n;
      out[(size_t)row * CC + col] = acc[s][r];
    }
  }
}

extern "C" void kernel_launch(void* const* d_in, const int* in_sizes, int n_in,
                              void* d_out, int out_size, void* d_ws, size_t ws_size,
                              hipStream_t stream) {
  (void)in_sizes; (void)n_in; (void)out_size; (void)ws_size;
  const float* x    = (const float*)d_in[0];
  const float* wqkv = (const float*)d_in[1];
  const float* wout = (const float*)d_in[2];
  float* out = (float*)d_out;

  char* ws = (char*)d_ws;
  size_t off = 0;
  u16* xb    = (u16*)(ws + off); off += (size_t)4 * TT * CC * 2;       // x bf16
  u16* wqkvb = (u16*)(ws + off); off += (size_t)3 * CC * CC * 2;       // w_qkv bf16
  u16* woutb = (u16*)(ws + off); off += (size_t)CC * CC * 2;           // w_out bf16
  u16* Qd    = (u16*)(ws + off); off += (size_t)4 * NHH * TT * HD * 2; // Q (roped)
  u16* Kd    = (u16*)(ws + off); off += (size_t)4 * NHH * TT * HD * 2; // K (roped)
  u16* Vd    = (u16*)(ws + off); off += (size_t)4 * NHH * TT * HD * 2; // V transposed
  u16* Od    = (u16*)(ws + off); off += (size_t)4 * TT * CC * 2;       // attn out bf16

  k_convert<<<1024, 256, 0, stream>>>(x,    xb,    4 * TT * CC);
  k_convert<<<512,  256, 0, stream>>>(wqkv, wqkvb, 3 * CC * CC);
  k_convert<<<256,  256, 0, stream>>>(wout, woutb, CC * CC);

  // 512 row-tiles x 18 col-tiles = 9216 waves, 4 waves/block
  k_qkv <<<2304, 128, 0, stream>>>(xb, wqkvb, Qd, Kd, Vd);
  // 4*12*128 = 6144 q-tiles
  k_attn<<<1536, 128, 0, stream>>>(Qd, Kd, Vd, Od);
  // 512 x 6 = 3072 waves
  k_proj<<<768, 128, 0, stream>>>(Od, woutb, out);
}
